// EfficientAttention_51092930953604
// MI455X (gfx1250) — matrix operand order repaired and checked by reading, compiled, not run on hardware
//
#include <hip/hip_runtime.h>
#include <math.h>

// ---------------------------------------------------------------------------
// Efficient attention on MI455X (gfx1250, wave32).
// All 6 GEMMs on v_wmma_f32_16x16x32_bf16 (f32 accumulate), softmaxes in f32.
// Fragment layouts (per CDNA5 ISA):
//   A frag: lane&15 = M row, K chunks at (lane>>4)*8 and +16  (2 x 16B loads)
//   B frag: lane&15 = N col, K chunks at (lane>>4)*8 and +16  (2 x 16B loads)
//   C/D  : VGPR r -> M = r + 8*(lane>>4), N = lane&15
// Round 3: the register double-buffered GEMMs (ctx_partial, final) move to
// 8 waves x 2 B-tiles per wave (was 4 waves x 4) so the cur/next fragment
// buffers fit in VGPRs and the steady-state loop keeps loads in flight
// behind the WMMAs instead of degenerating to s_wait_loadcnt 0.
// Sizes hardcoded from setup_inputs: n=8, L=4096, C=512, H=8, hd=64.
// ---------------------------------------------------------------------------

typedef __attribute__((ext_vector_type(16))) __bf16 v16bf;
typedef __attribute__((ext_vector_type(8)))  __bf16 v8bf;
typedef __attribute__((ext_vector_type(8)))  float  v8f;
typedef __attribute__((ext_vector_type(8)))  unsigned short v8us;

#define NB 8
#define LL 4096
#define CC 512
#define HH 8
#define HD 64
#define LPARTS 8            // L-split for ctx reduction
#define LSLICE (LL / LPARTS)

__device__ __forceinline__ unsigned short f2bf_bits(float f) {
  unsigned int x; __builtin_memcpy(&x, &f, 4);
  unsigned int r = x + 0x7FFFu + ((x >> 16) & 1u);   // round-to-nearest-even
  return (unsigned short)(r >> 16);
}
__device__ __forceinline__ __bf16 f2bf(float f) {
  unsigned short s = f2bf_bits(f);
  __bf16 b; __builtin_memcpy(&b, &s, 2); return b;
}

// A fragment: rowptr points at K=0 of this lane's M row (row has >=32 bf16).
__device__ __forceinline__ v16bf load_a_bf(const __bf16* rowptr, int lane) {
  const int kb = (lane >> 4) * 8;
  v8bf lo = *(const v8bf*)(rowptr + kb);
  v8bf hi = *(const v8bf*)(rowptr + kb + 16);
  return __builtin_shufflevector(lo, hi, 0,1,2,3,4,5,6,7,8,9,10,11,12,13,14,15);
}
// B fragment: colptr points at K=0 of this lane's N column; K chunks at (lane>>4)*8 and +16.
__device__ __forceinline__ v16bf load_b_bf(const __bf16* colptr, int lane) {
  const int kb = (lane >> 4) * 8;
  v8bf lo = *(const v8bf*)(colptr + kb);
  v8bf hi = *(const v8bf*)(colptr + kb + 16);
  return __builtin_shufflevector(lo, hi, 0,1,2,3,4,5,6,7,8,9,10,11,12,13,14,15);
}
// A fragment from f32 source, converted to bf16 on the fly.
__device__ __forceinline__ v16bf load_a_f32(const float* rowptr, int lane) {
  const int kb = (lane >> 4) * 8;
  v16bf r;
#pragma unroll
  for (int i = 0; i < 8; ++i) { r[i] = f2bf(rowptr[kb+i]); r[i+8] = f2bf(rowptr[kb+16+i]); }
  return r;
}
// Store one D accumulator (8 f32 for 8 consecutive M) as 8 contiguous bf16.
__device__ __forceinline__ void store8_bf(__bf16* p, v8f d) {
  v8us u;
#pragma unroll
  for (int i = 0; i < 8; ++i) u[i] = f2bf_bits(d[i]);
  *(v8us*)p = u;
}
__device__ __forceinline__ v8f wmma_bf16(v16bf a, v16bf b, v8f c) {
  return __builtin_amdgcn_wmma_f32_16x16x32_bf16(false, a, false, b, (short)0, c,
                                                 false, false);
}

// ---------------------------------------------------------------------------
// Projection GEMM: Y[k,l] = sum_c W[k,c] * src[n,l,c] + bias[k]
// 4 waves / WG, 64(k) x 64(l) tile, K staged through LDS as bf16 with a
// software pipeline: chunk i+1 is pulled from global into registers while
// chunk i's WMMAs run, then written to LDS after the read barrier.
// MODE 0: q -> f32 [n][l][ck]   (ck contiguous; per-head softmax later)
// MODE 1: k -> f32 [n][ck][l]   (l contiguous; row softmax later)
// MODE 2: v -> bf16 [n][cv][l]  (l contiguous; consumed by ctx GEMM)
// ---------------------------------------------------------------------------
template<int MODE>
__global__ __launch_bounds__(128)
void proj_kernel(const float* __restrict__ W, const float* __restrict__ bias,
                 const float* __restrict__ src, float* __restrict__ outf,
                 __bf16* __restrict__ outb) {
  __shared__ __bf16 Wl[64 * 32];
  __shared__ __bf16 Xl[64 * 32];
  const int tid = threadIdx.x, lane = tid & 31, wave = tid >> 5;
  const int nl = lane & 15, hi = lane >> 4;
  const int lb = blockIdx.x * 64, kb = blockIdx.y * 64, nb = blockIdx.z;

  v8f zero = {0.f,0.f,0.f,0.f,0.f,0.f,0.f,0.f};
  v8f acc[4];
#pragma unroll
  for (int t = 0; t < 4; ++t) acc[t] = zero;

  const float* srcBase = src + ((size_t)nb * LL + lb) * CC;
  const float* Wbase   = W + (size_t)kb * CC;

  // stage chunk 0
#pragma unroll
  for (int j = 0; j < 16; ++j) {
    int e = tid + 128 * j;
    int row = e >> 5, col = e & 31;
    Wl[e] = f2bf(Wbase[(size_t)row * CC + col]);
    Xl[e] = f2bf(srcBase[(size_t)row * CC + col]);
  }
  __syncthreads();

  float wreg[16], xreg[16];
  for (int it = 0; it < 16; ++it) {
    const int cnext = (it + 1) * 32;
    if (it < 15) {                         // global loads for chunk i+1 in
#pragma unroll                             // flight during chunk i's WMMAs
      for (int j = 0; j < 16; ++j) {
        int e = tid + 128 * j;
        int row = e >> 5, col = e & 31;
        wreg[j] = Wbase[(size_t)row * CC + cnext + col];
        xreg[j] = srcBase[(size_t)row * CC + cnext + col];
      }
    }
    v16bf a = load_a_bf(Wl + (16 * wave + nl) * 32, lane);
#pragma unroll
    for (int t = 0; t < 4; ++t) {
      v16bf b = load_b_bf(Xl + (16 * t + nl) * 32, lane);
      acc[t] = wmma_bf16(a, b, acc[t]);
    }
    __syncthreads();                       // done reading chunk i
    if (it < 15) {
#pragma unroll
      for (int j = 0; j < 16; ++j) {
        int e = tid + 128 * j;
        Wl[e] = f2bf(wreg[j]);
        Xl[e] = f2bf(xreg[j]);
      }
    }
    __syncthreads();                       // chunk i+1 visible
  }

  const int kg0 = kb + 16 * wave + 8 * hi;  // 8 consecutive M rows
  v8f bvec = *(const v8f*)(bias + kg0);
#pragma unroll
  for (int t = 0; t < 4; ++t) {
    v8f d = acc[t] + bvec;
    const int l = lb + 16 * t + nl;
    if (MODE == 0) {
      *(v8f*)(outf + ((size_t)nb * LL + l) * CC + kg0) = d;   // [l][ck], 32B
    } else if (MODE == 1) {
#pragma unroll
      for (int r = 0; r < 8; ++r)
        outf[((size_t)nb * CC + kg0 + r) * LL + l] = d[r];    // [ck][l]
    } else {
#pragma unroll
      for (int r = 0; r < 8; ++r)
        outb[((size_t)nb * CC + kg0 + r) * LL + l] = f2bf(d[r]);
    }
  }
}

// ---------------------------------------------------------------------------
// Row softmax over L=4096 (k projection).  One block per (n,ck) row.
// ---------------------------------------------------------------------------
__global__ __launch_bounds__(256)
void ksoftmax_kernel(const float* __restrict__ kp, __bf16* __restrict__ ksm) {
  const size_t row = blockIdx.x;
  const float* src = kp + row * LL;
  __bf16* dst = ksm + row * LL;
  const int tid = threadIdx.x, lane = tid & 31, wave = tid >> 5;
  __shared__ float red[8];

  float m = -3.0e38f;
  for (int i = tid; i < LL; i += 256) m = fmaxf(m, src[i]);
#pragma unroll
  for (int o = 16; o > 0; o >>= 1) m = fmaxf(m, __shfl_xor(m, o, 32));
  if (lane == 0) red[wave] = m;
  __syncthreads();
  float bm = red[0];
#pragma unroll
  for (int j = 1; j < 8; ++j) bm = fmaxf(bm, red[j]);
  __syncthreads();

  float s = 0.f;
  for (int i = tid; i < LL; i += 256) s += __expf(src[i] - bm);
#pragma unroll
  for (int o = 16; o > 0; o >>= 1) s += __shfl_xor(s, o, 32);
  if (lane == 0) red[wave] = s;
  __syncthreads();
  float bs = 0.f;
#pragma unroll
  for (int j = 0; j < 8; ++j) bs += red[j];
  const float inv = 1.0f / bs;

  for (int i = tid; i < LL; i += 256) dst[i] = f2bf(__expf(src[i] - bm) * inv);
}

// ---------------------------------------------------------------------------
// Per-(n,l,head) softmax over 64 contiguous channels (q projection).
// 8 waves / block, one 64-value segment per wave (2 values per lane).
// ---------------------------------------------------------------------------
__global__ __launch_bounds__(256)
void qsoftmax_kernel(const float* __restrict__ qp, __bf16* __restrict__ qsm) {
  const int lane = threadIdx.x & 31, wave = threadIdx.x >> 5;
  const size_t seg = (size_t)blockIdx.x * 8 + wave;   // (n*L*H) segments
  const float* src = qp + seg * 64;
  __bf16* dst = qsm + seg * 64;
  float a = src[lane], b = src[lane + 32];
  float m = fmaxf(a, b);
#pragma unroll
  for (int o = 16; o > 0; o >>= 1) m = fmaxf(m, __shfl_xor(m, o, 32));
  float e0 = __expf(a - m), e1 = __expf(b - m);
  float s = e0 + e1;
#pragma unroll
  for (int o = 16; o > 0; o >>= 1) s += __shfl_xor(s, o, 32);
  const float inv = 1.f / s;
  dst[lane]      = f2bf(e0 * inv);
  dst[lane + 32] = f2bf(e1 * inv);
}

// ---------------------------------------------------------------------------
// ctx partials: ctx_p[kk,v] = sum_{l in slice} ksm[kk,l] * val[v,l]
// per (n,head,slice).  8 slices of L=512 -> 512 WGs.
// 8 waves / WG; wave owns one 16-kk strip and TWO 16-v B tiles so the
// cur/next double buffers (A 16 + B 32 + acc 16 VGPRs) stay register-resident.
// Partials stored f32 transposed [v][kk] (kk contiguous).
// ---------------------------------------------------------------------------
__global__ __launch_bounds__(256)
void ctx_partial_kernel(const __bf16* __restrict__ ksm,
                        const __bf16* __restrict__ valb,
                        float* __restrict__ ctxp) {
  const int part = blockIdx.x;
  const int nh = blockIdx.y;
  const int nb = nh / HH, h = nh % HH;
  const int tid = threadIdx.x, lane = tid & 31, wave = tid >> 5;
  const int nl = lane & 15, hi = lane >> 4;
  const int ks = (wave >> 1) * 16;     // kk strip for this wave
  const int vs = (wave & 1) * 32;      // base of this wave's two v tiles

  const __bf16* kbase = ksm  + ((size_t)nb * CC + h * HD) * LL + part * LSLICE;
  const __bf16* vbase = valb + ((size_t)nb * CC + h * HD) * LL + part * LSLICE;
  const __bf16* arow  = kbase + (size_t)(ks + nl) * LL;
  const __bf16* brow0 = vbase + (size_t)(vs + nl) * LL;
  const __bf16* brow1 = vbase + (size_t)(vs + 16 + nl) * LL;

  v8f zero = {0.f,0.f,0.f,0.f,0.f,0.f,0.f,0.f};
  v8f acc0 = zero, acc1 = zero;

  v16bf a_cur  = load_a_bf(arow, lane);
  v16bf b0_cur = load_b_bf(brow0, lane);
  v16bf b1_cur = load_b_bf(brow1, lane);

#pragma unroll
  for (int it = 0; it < LSLICE / 32; ++it) {
    v16bf a_nxt = a_cur, b0_nxt = b0_cur, b1_nxt = b1_cur;
    if (it < LSLICE / 32 - 1) {            // chunk i+1 loads in flight
      const int l0n = (it + 1) * 32;
      a_nxt  = load_a_bf(arow + l0n, lane);
      b0_nxt = load_b_bf(brow0 + l0n, lane);
      b1_nxt = load_b_bf(brow1 + l0n, lane);
    }
    acc0 = wmma_bf16(a_cur, b0_cur, acc0);
    acc1 = wmma_bf16(a_cur, b1_cur, acc1);
    a_cur = a_nxt; b0_cur = b0_nxt; b1_cur = b1_nxt;
  }

  const int kk0 = ks + 8 * hi;
  float* obase = ctxp + (size_t)(nh * LPARTS + part) * (HD * HD);
  *(v8f*)(obase + (size_t)(vs + nl) * HD + kk0)      = acc0;
  *(v8f*)(obase + (size_t)(vs + 16 + nl) * HD + kk0) = acc1;
}

// Sum the 8 slice partials, emit bf16 ctxb[nh][v][kk].
__global__ __launch_bounds__(256)
void ctx_reduce_kernel(const float* __restrict__ ctxp, __bf16* __restrict__ ctxb) {
  const int idx = blockIdx.x * 256 + threadIdx.x;   // over nh*64*64
  const int nh = idx >> 12, rem = idx & 4095;
  float s = 0.f;
#pragma unroll
  for (int p = 0; p < LPARTS; ++p)
    s += ctxp[((size_t)nh * LPARTS + p) * (HD * HD) + rem];
  ctxb[(size_t)nh * (HD * HD) + rem] = f2bf(s);
}

// ---------------------------------------------------------------------------
// att[v,l] = sum_kk ctx[kk,v] * qsm[kk,l]; stored as aggb[n][l][cv]
// (cv contiguous) so the final GEMM's B fragments are contiguous.
// K=64 -> fully unrolled, all loads hoistable ahead of the WMMAs.
// ---------------------------------------------------------------------------
__global__ __launch_bounds__(128)
void att_kernel(const __bf16* __restrict__ ctxb, const __bf16* __restrict__ qsm,
                __bf16* __restrict__ aggb) {
  const int nh = blockIdx.y, nb = nh / HH, h = nh % HH;
  const int l0 = blockIdx.x * 64;
  const int lane = threadIdx.x & 31, wave = threadIdx.x >> 5;
  const int nl = lane & 15, hi = lane >> 4;
  const __bf16* cb = ctxb + (size_t)nh * HD * HD;

  v8f zero = {0.f,0.f,0.f,0.f,0.f,0.f,0.f,0.f};
  v8f acc[4];
#pragma unroll
  for (int t = 0; t < 4; ++t) acc[t] = zero;

#pragma unroll
  for (int k0 = 0; k0 < HD; k0 += 32) {
    v16bf a = load_a_bf(cb + (size_t)(16 * wave + nl) * HD + k0, lane); // rows v
#pragma unroll
    for (int t = 0; t < 4; ++t) {
      const int l = l0 + 16 * t + nl;
      v16bf b = load_b_bf(qsm + ((size_t)nb * LL + l) * CC + h * HD + k0, lane);
      acc[t] = wmma_bf16(a, b, acc[t]);
    }
  }
  const int vg0 = 16 * wave + 8 * hi;
#pragma unroll
  for (int t = 0; t < 4; ++t) {
    const int l = l0 + 16 * t + nl;
    store8_bf(aggb + ((size_t)nb * LL + l) * CC + h * HD + vg0, acc[t]);
  }
}

// ---------------------------------------------------------------------------
// out[n,c,l] = sum_v Wr[c,v] * agg[n,l,v] + br[c] + x[n,l,c]
// 8 waves / WG; wave owns one 16-c strip and TWO 16-l B tiles (same
// register-resident double-buffer scheme as ctx_partial).
// ---------------------------------------------------------------------------
__global__ __launch_bounds__(256)
void final_kernel(const float* __restrict__ Wr, const float* __restrict__ br,
                  const __bf16* __restrict__ aggb, const float* __restrict__ x,
                  float* __restrict__ out) {
  const int lb = blockIdx.x * 64, cb0 = blockIdx.y * 64, nb = blockIdx.z;
  const int tid = threadIdx.x, lane = tid & 31, wave = tid >> 5;
  const int nl = lane & 15, hi = lane >> 4;
  const int cs = (wave >> 1) * 16;     // c strip for this wave
  const int ls = (wave & 1) * 32;      // base of this wave's two l tiles

  const float* wrow   = Wr + (size_t)(cb0 + cs + nl) * CC;
  const __bf16* brow0 = aggb + ((size_t)nb * LL + lb + ls + nl) * CC;
  const __bf16* brow1 = aggb + ((size_t)nb * LL + lb + ls + 16 + nl) * CC;

  v8f zero = {0.f,0.f,0.f,0.f,0.f,0.f,0.f,0.f};
  v8f acc0 = zero, acc1 = zero;

  v16bf a_cur  = load_a_f32(wrow, lane);
  v16bf b0_cur = load_b_bf(brow0, lane);
  v16bf b1_cur = load_b_bf(brow1, lane);

#pragma unroll
  for (int it = 0; it < CC / 32; ++it) {
    v16bf a_nxt = a_cur, b0_nxt = b0_cur, b1_nxt = b1_cur;
    if (it < CC / 32 - 1) {
      const int v0n = (it + 1) * 32;
      a_nxt  = load_a_f32(wrow + v0n, lane);
      b0_nxt = load_b_bf(brow0 + v0n, lane);
      b1_nxt = load_b_bf(brow1 + v0n, lane);
    }
    acc0 = wmma_bf16(a_cur, b0_cur, acc0);
    acc1 = wmma_bf16(a_cur, b1_cur, acc1);
    a_cur = a_nxt; b0_cur = b0_nxt; b1_cur = b1_nxt;
  }

  const int cg0 = cb0 + cs + 8 * hi;
  v8f bvec = *(const v8f*)(br + cg0);
#pragma unroll
  for (int j = 0; j < 2; ++j) {
    const int l = lb + ls + 16 * j + nl;
    v8f xv = *(const v8f*)(x + ((size_t)nb * LL + l) * CC + cg0);
    v8f d = (j == 0 ? acc0 : acc1) + bvec + xv;
#pragma unroll
    for (int r = 0; r < 8; ++r)
      out[((size_t)nb * CC + cg0 + r) * LL + l] = d[r];
  }
}

// ---------------------------------------------------------------------------
extern "C" void kernel_launch(void* const* d_in, const int* in_sizes, int n_in,
                              void* d_out, int out_size, void* d_ws, size_t ws_size,
                              hipStream_t stream) {
  const float* x  = (const float*)d_in[0];
  const float* v  = (const float*)d_in[1];
  const float* Wq = (const float*)d_in[2];
  const float* bq = (const float*)d_in[3];
  const float* Wk = (const float*)d_in[4];
  const float* bk = (const float*)d_in[5];
  const float* Wv = (const float*)d_in[6];
  const float* bv = (const float*)d_in[7];
  const float* Wr = (const float*)d_in[8];
  const float* br = (const float*)d_in[9];
  float* out = (float*)d_out;

  char* ws = (char*)d_ws;
  const size_t SZ_F32 = (size_t)NB * CC * LL * 4;  // 64 MB
  const size_t SZ_BF  = (size_t)NB * CC * LL * 2;  // 32 MB
  const size_t SZ_CTXP = (size_t)NB * HH * LPARTS * HD * HD * 4;  // 8 MB
  float*  qp   = (float*)(ws + 0);                       // q proj f32 [n][l][ck]
  float*  kp   = (float*)(ws + SZ_F32);                  // k proj f32 [n][ck][l]
  __bf16* valb = (__bf16*)(ws + 2 * SZ_F32);             // v proj bf16 [n][cv][l]
  __bf16* ksm  = (__bf16*)(ws + 2 * SZ_F32 + SZ_BF);     // softmax(k) bf16
  __bf16* qsm  = (__bf16*)(ws + 2 * SZ_F32 + 2 * SZ_BF); // softmax(q) bf16
  float*  ctxp = (float*)(ws + SZ_F32);                  // alias kp (dead after ksoftmax)
  __bf16* ctxb = (__bf16*)(ws + SZ_F32 + SZ_CTXP);       // alias kp region too
  __bf16* aggb = (__bf16*)(ws + 0);                      // alias qp (dead after qsoftmax)

  dim3 gProj(LL / 64, CC / 64, NB);
  proj_kernel<0><<<gProj, 128, 0, stream>>>(Wq, bq, x, qp, nullptr);
  proj_kernel<1><<<gProj, 128, 0, stream>>>(Wk, bk, v, kp, nullptr);
  proj_kernel<2><<<gProj, 128, 0, stream>>>(Wv, bv, v, nullptr, valb);

  ksoftmax_kernel<<<NB * CC, 256, 0, stream>>>(kp, ksm);
  qsoftmax_kernel<<<NB * LL * HH / 8, 256, 0, stream>>>(qp, qsm);

  ctx_partial_kernel<<<dim3(LPARTS, NB * HH), 256, 0, stream>>>(ksm, valb, ctxp);
  ctx_reduce_kernel<<<(NB * HH * HD * HD) / 256, 256, 0, stream>>>(ctxp, ctxb);

  att_kernel<<<dim3(LL / 64, NB * HH), 128, 0, stream>>>(ctxb, qsm, aggb);
  final_kernel<<<dim3(LL / 64, CC / 64, NB), 256, 0, stream>>>(Wr, br, aggb, x, out);
}
